// SBERTaAttention_1443109011848
// MI455X (gfx1250) — matrix operand order repaired and checked by reading, not compiled, not run on hardware
//
#include <hip/hip_runtime.h>

// ---------------------------------------------------------------------------
// SBERTa attention for MI455X (gfx1250): bf16 WMMA everywhere, flash-style
// online-softmax attention (scores never hit memory), wave32 layouts.
// V is pre-transposed once into [B,H,DH,T] so the flash loop stages both K
// and V^T tiles with pure b128 traffic.
// ---------------------------------------------------------------------------

typedef __attribute__((ext_vector_type(16))) __bf16       v16bf;
typedef __attribute__((ext_vector_type(8)))  float        v8f;
typedef __attribute__((ext_vector_type(4)))  unsigned int u32x4;
typedef __attribute__((ext_vector_type(2)))  unsigned int u32x2;
typedef __attribute__((ext_vector_type(4)))  float        f32x4;

#define DEV __device__ __forceinline__

constexpr int Bc  = 2;
constexpr int Tc  = 2048;
constexpr int Dc  = 1024;
constexpr int Hc  = 16;
constexpr int DHc = 64;
constexpr int Kc  = 8;          // language-compat dim
constexpr int Mtot = Bc * Tc;   // 4096 rows

DEV unsigned short f2bf(float f) {
  unsigned int u = __float_as_uint(f);
  u += 0x7fffu + ((u >> 16) & 1u);           // round-to-nearest-even
  return (unsigned short)(u >> 16);
}

union FragU { u32x4 u[2]; v16bf v; };

// A-matrix 16x32 bf16 fragment (lane&15 = M row; lanes 0-15 hold K 0-7 &
// 16-23, lanes 16-31 hold K 8-15 & 24-31).  p0 points at (row0, k0).
DEV v16bf load_a(const unsigned short* p0, int ld, int lane) {
  const unsigned short* p = p0 + (size_t)(lane & 15) * ld + ((lane >> 4) << 3);
  FragU r;
  r.u[0] = *(const u32x4*)(p);
  r.u[1] = *(const u32x4*)(p + 16);
  return r.v;
}

// B-matrix 32x16 bf16 fragment from a row-major [N][K] source (lane&15 = N;
// lanes 0-15 hold K 0-7 & 16-23, lanes 16-31 hold K 8-15 & 24-31).
DEV v16bf load_b(const unsigned short* p0, int ld, int lane) {
  const unsigned short* p = p0 + (size_t)(lane & 15) * ld + ((lane >> 4) << 3);
  FragU r;
  r.u[0] = *(const u32x4*)(p);
  r.u[1] = *(const u32x4*)(p + 16);
  return r.v;
}

DEV v8f wmma_bf16(v16bf a, v16bf b, v8f c) {
  return __builtin_amdgcn_wmma_f32_16x16x32_bf16(
      /*neg_a=*/false, a, /*neg_b=*/false, b,
      /*c_mod=*/(short)0, c, /*reuse_a=*/false, /*reuse_b=*/false);
}

// ---------------------------------------------------------------------------
// f32 -> bf16 convert, 4 elements / thread
// ---------------------------------------------------------------------------
__global__ __launch_bounds__(256)
void cvt_bf16x4(const float* __restrict__ in, unsigned short* __restrict__ out,
                int n4) {
  int i = blockIdx.x * blockDim.x + threadIdx.x;
  if (i >= n4) return;
  f32x4 f = ((const f32x4*)in)[i];
  u32x2 r;
  r.x = (unsigned int)f2bf(f.x) | ((unsigned int)f2bf(f.y) << 16);
  r.y = (unsigned int)f2bf(f.z) | ((unsigned int)f2bf(f.w) << 16);
  ((u32x2*)out)[i] = r;
}

// ---------------------------------------------------------------------------
// p_compat[b,h,t,l] = sum_k p[b,t,k] * compat[h,k,l]
// ---------------------------------------------------------------------------
__global__ __launch_bounds__(256)
void pc_kernel(const float* __restrict__ p, const float* __restrict__ compat,
               float* __restrict__ pc) {
  int idx = blockIdx.x * blockDim.x + threadIdx.x;   // 2^19 total
  int l = idx & 7;
  int t = (idx >> 3) & (Tc - 1);
  int h = (idx >> 14) & (Hc - 1);
  int b = idx >> 18;
  const float* pr = p + ((size_t)b * Tc + t) * Kc;
  const float* cr = compat + (size_t)h * Kc * Kc + l;
  float acc = 0.f;
#pragma unroll
  for (int k = 0; k < Kc; ++k) acc += pr[k] * cr[k * Kc];
  pc[idx] = acc;
}

// ---------------------------------------------------------------------------
// QKV projection: O = X * W^T  (X, W, O all bf16, K=1024)
// block = 256 threads (8 waves); block tile 128x128; wave tile 32x64.
// grid.z picks which of Wq/Wk/Wv.
// ---------------------------------------------------------------------------
__global__ __launch_bounds__(256)
void gemm_qkv(const unsigned short* __restrict__ X,
              const unsigned short* __restrict__ W0,
              const unsigned short* __restrict__ W1,
              const unsigned short* __restrict__ W2,
              unsigned short* __restrict__ O0,
              unsigned short* __restrict__ O1,
              unsigned short* __restrict__ O2) {
  const unsigned short* W = (blockIdx.z == 0) ? W0 : (blockIdx.z == 1 ? W1 : W2);
  unsigned short*       O = (blockIdx.z == 0) ? O0 : (blockIdx.z == 1 ? O1 : O2);

  const int lane = threadIdx.x & 31;
  const int wave = threadIdx.x >> 5;
  const int m0 = blockIdx.y * 128 + (wave & 3) * 32;
  const int n0 = blockIdx.x * 128 + (wave >> 2) * 64;

  v8f c[2][4];
#pragma unroll
  for (int i = 0; i < 2; ++i)
#pragma unroll
    for (int j = 0; j < 4; ++j) c[i][j] = (v8f)0.f;

  for (int k0 = 0; k0 < Dc; k0 += 32) {
    __builtin_prefetch(X + (size_t)m0 * Dc + k0 + 128, 0, 3);
    v16bf a0 = load_a(X + (size_t)(m0)      * Dc + k0, Dc, lane);
    v16bf a1 = load_a(X + (size_t)(m0 + 16) * Dc + k0, Dc, lane);
    v16bf b0 = load_b(W + (size_t)(n0)      * Dc + k0, Dc, lane);
    v16bf b1 = load_b(W + (size_t)(n0 + 16) * Dc + k0, Dc, lane);
    v16bf b2 = load_b(W + (size_t)(n0 + 32) * Dc + k0, Dc, lane);
    v16bf b3 = load_b(W + (size_t)(n0 + 48) * Dc + k0, Dc, lane);
    c[0][0] = wmma_bf16(a0, b0, c[0][0]);
    c[0][1] = wmma_bf16(a0, b1, c[0][1]);
    c[0][2] = wmma_bf16(a0, b2, c[0][2]);
    c[0][3] = wmma_bf16(a0, b3, c[0][3]);
    c[1][0] = wmma_bf16(a1, b0, c[1][0]);
    c[1][1] = wmma_bf16(a1, b1, c[1][1]);
    c[1][2] = wmma_bf16(a1, b2, c[1][2]);
    c[1][3] = wmma_bf16(a1, b3, c[1][3]);
  }

  const int cr = (lane >> 4) * 8;
  const int cc = lane & 15;
#pragma unroll
  for (int i = 0; i < 2; ++i)
#pragma unroll
    for (int j = 0; j < 4; ++j)
#pragma unroll
      for (int e = 0; e < 8; ++e) {
        int m = m0 + i * 16 + cr + e;
        int n = n0 + j * 16 + cc;
        O[(size_t)m * Dc + n] = f2bf(c[i][j][e]);
      }
}

// ---------------------------------------------------------------------------
// Output projection: out = ctx * Wo^T + bo   (ctx,Wo bf16; out f32)
// ---------------------------------------------------------------------------
__global__ __launch_bounds__(256)
void gemm_out(const unsigned short* __restrict__ X,
              const unsigned short* __restrict__ W,
              const float* __restrict__ bias,
              float* __restrict__ O) {
  const int lane = threadIdx.x & 31;
  const int wave = threadIdx.x >> 5;
  const int m0 = blockIdx.y * 128 + (wave & 3) * 32;
  const int n0 = blockIdx.x * 128 + (wave >> 2) * 64;

  v8f c[2][4];
#pragma unroll
  for (int i = 0; i < 2; ++i)
#pragma unroll
    for (int j = 0; j < 4; ++j) c[i][j] = (v8f)0.f;

  for (int k0 = 0; k0 < Dc; k0 += 32) {
    __builtin_prefetch(X + (size_t)m0 * Dc + k0 + 128, 0, 3);
    v16bf a0 = load_a(X + (size_t)(m0)      * Dc + k0, Dc, lane);
    v16bf a1 = load_a(X + (size_t)(m0 + 16) * Dc + k0, Dc, lane);
#pragma unroll
    for (int j = 0; j < 4; ++j) {
      v16bf bj = load_b(W + (size_t)(n0 + 16 * j) * Dc + k0, Dc, lane);
      c[0][j] = wmma_bf16(a0, bj, c[0][j]);
      c[1][j] = wmma_bf16(a1, bj, c[1][j]);
    }
  }

  const int cr = (lane >> 4) * 8;
  const int cc = lane & 15;
#pragma unroll
  for (int i = 0; i < 2; ++i)
#pragma unroll
    for (int j = 0; j < 4; ++j) {
      int n = n0 + j * 16 + cc;
      float bn = bias[n];
#pragma unroll
      for (int e = 0; e < 8; ++e) {
        int m = m0 + i * 16 + cr + e;
        O[(size_t)m * Dc + n] = c[i][j][e] + bn;
      }
    }
}

// ---------------------------------------------------------------------------
// One-time V transpose: Vb [B*T, D] bf16 -> Vt [B,H,DH,T] bf16.
// 64x64 tile per block through padded LDS; both sides fully coalesced b128.
// grid = (T/64, H, B), block = 256.
// ---------------------------------------------------------------------------
__global__ __launch_bounds__(256)
void vtrans_kernel(const unsigned short* __restrict__ Vb,
                   unsigned short* __restrict__ Vt) {
  __shared__ unsigned short tile[64][72];    // 72-ushort stride: 16B aligned,
                                             // and 36-bank step on col reads
  const int tid = threadIdx.x;
  const int tt = blockIdx.x, h = blockIdx.y, b = blockIdx.z;
  const int t0 = tt * 64;

  // read phase: rows = t, cols = d
  {
    int r  = tid >> 2;            // 0..63 (t offset)
    int cs = (tid & 3) * 16;      // 0,16,32,48 (d offset)
    const unsigned short* src =
        Vb + ((size_t)b * Tc + t0 + r) * Dc + h * DHc + cs;
    *(u32x4*)&tile[r][cs]     = *(const u32x4*)(src);
    *(u32x4*)&tile[r][cs + 8] = *(const u32x4*)(src + 8);
  }
  __syncthreads();
  // write phase: rows = d, cols = t
  {
    int d  = tid >> 2;            // 0..63 (d offset)
    int cs = (tid & 3) * 16;      // 0,16,32,48 (t offset)
    alignas(16) unsigned short tmp[16];
#pragma unroll
    for (int q = 0; q < 16; ++q) tmp[q] = tile[cs + q][d];
    unsigned short* dst =
        Vt + (((size_t)b * Hc + h) * DHc + d) * Tc + t0 + cs;
    *(u32x4*)(dst)     = *(u32x4*)(tmp);
    *(u32x4*)(dst + 8) = *(u32x4*)(tmp + 8);
  }
}

// ---------------------------------------------------------------------------
// Flash attention.  grid = (T/128, H, B), block = 256 (8 waves).
// Wave w owns query rows [qt*128 + w*16, +16).  Key loop in tiles of 64.
// ---------------------------------------------------------------------------
__global__ __launch_bounds__(256)
void attn_kernel(const unsigned short* __restrict__ Q,
                 const unsigned short* __restrict__ Km,
                 const unsigned short* __restrict__ Vtg,  // [B,H,DH,T] bf16
                 const float* __restrict__ pc,     // [B,H,T,8]
                 const float* __restrict__ p,      // [B,T,8]
                 const float* __restrict__ s,      // [B,T]
                 const float* __restrict__ mask,   // [B,1,1,T]
                 const float* __restrict__ gamma,  // [1]
                 unsigned short* __restrict__ ctx) // [B*T, D] bf16
{
  __shared__ unsigned short Ks[64 * 64];     // key tile, row-major [j][d]
  __shared__ unsigned short Vt[64 * 64];     // value tile transposed [d][j]
  __shared__ unsigned short Ps[8][16 * 64];  // per-wave P tile (bf16)
  __shared__ float pK[64 * 8];               // p rows for key tile
  __shared__ float kb[64];                   // gamma*s + mask per key col

  const int tid  = threadIdx.x;
  const int lane = tid & 31;
  const int wave = tid >> 5;
  const int qt = blockIdx.x, h = blockIdx.y, b = blockIdx.z;
  const float g = gamma[0];
  const float rscale = 0.125f;               // 1/sqrt(64)

  const size_t rowQ = (size_t)b * Tc + qt * 128 + wave * 16;  // first q row

  // Q fragments live in registers for the whole kernel (DH=64 -> 2 k-steps)
  v16bf qa0 = load_a(Q + rowQ * Dc + h * DHc + 0,  Dc, lane);
  v16bf qa1 = load_a(Q + rowQ * Dc + h * DHc + 32, Dc, lane);

  // per-lane p_compat rows for the 8 rows this lane-half touches
  float pcr[8][8];
  {
    const size_t base =
        (((size_t)b * Hc + h) * Tc + qt * 128 + wave * 16 + (lane >> 4) * 8) * Kc;
#pragma unroll
    for (int e = 0; e < 8; ++e)
#pragma unroll
      for (int l = 0; l < Kc; ++l) pcr[e][l] = pc[base + (size_t)e * Kc + l];
  }

  float mrow[8], lrow[8];
  v8f acc[4];
#pragma unroll
  for (int e = 0; e < 8; ++e) { mrow[e] = -1e30f; lrow[e] = 0.f; }
#pragma unroll
  for (int n = 0; n < 4; ++n) acc[n] = (v8f)0.f;

  for (int j0 = 0; j0 < Tc; j0 += 64) {
    __syncthreads();   // previous iteration's WMMA reads of Ks/Vt are done
    {
      int r  = tid >> 2;            // 0..63
      int cs = (tid & 3) * 16;      // 0,16,32,48
      // stage K tile: 64 j-rows x 64 d-cols bf16 (8 KB), 32 B per thread
      const unsigned short* src =
          Km + ((size_t)b * Tc + j0 + r) * Dc + h * DHc + cs;
      *(u32x4*)&Ks[r * 64 + cs]     = *(const u32x4*)(src);
      *(u32x4*)&Ks[r * 64 + cs + 8] = *(const u32x4*)(src + 8);
      // stage V^T tile: 64 d-rows x 64 j-cols, straight copy (pre-transposed)
      const unsigned short* vs =
          Vtg + (((size_t)b * Hc + h) * DHc + r) * Tc + j0 + cs;
      *(u32x4*)&Vt[r * 64 + cs]     = *(const u32x4*)(vs);
      *(u32x4*)&Vt[r * 64 + cs + 8] = *(const u32x4*)(vs + 8);
    }
    if (tid < 128) {   // 64 rows of p (8 f32 each) = 512 floats
      *(f32x4*)&pK[tid * 4] =
          *(const f32x4*)(p + ((size_t)b * Tc + j0) * Kc + tid * 4);
    }
    if (tid < 64) {
      kb[tid] = g * s[(size_t)b * Tc + j0 + tid] + mask[(size_t)b * Tc + j0 + tid];
    }
    __syncthreads();

    // ---- S = Q K^T (16x64 per wave) ----
    v8f S[4];
#pragma unroll
    for (int n = 0; n < 4; ++n) S[n] = (v8f)0.f;
#pragma unroll
    for (int ks = 0; ks < 2; ++ks) {
      v16bf a = ks ? qa1 : qa0;
#pragma unroll
      for (int n = 0; n < 4; ++n) {
        v16bf bk = load_b(&Ks[(n * 16) * 64 + ks * 32], 64, lane);
        S[n] = wmma_bf16(a, bk, S[n]);
      }
    }

    // ---- scale + language-compat bias + gamma*s + mask ----
#pragma unroll
    for (int n = 0; n < 4; ++n) {
      int j = n * 16 + (lane & 15);
      float pj[8];
#pragma unroll
      for (int l = 0; l < Kc; ++l) pj[l] = pK[j * Kc + l];
      float kbj = kb[j];
#pragma unroll
      for (int e = 0; e < 8; ++e) {
        float bias = kbj;
#pragma unroll
        for (int l = 0; l < Kc; ++l) bias += pcr[e][l] * pj[l];
        S[n][e] = S[n][e] * rscale + bias;
      }
    }

    // ---- online softmax (rows live in 16-lane halves) ----
#pragma unroll
    for (int e = 0; e < 8; ++e) {
      float mt = fmaxf(fmaxf(S[0][e], S[1][e]), fmaxf(S[2][e], S[3][e]));
#pragma unroll
      for (int o = 8; o >= 1; o >>= 1) mt = fmaxf(mt, __shfl_xor(mt, o, 16));
      float mnew  = fmaxf(mrow[e], mt);
      float alpha = __expf(mrow[e] - mnew);
      mrow[e] = mnew;
      float rs = 0.f;
#pragma unroll
      for (int n = 0; n < 4; ++n) {
        S[n][e] = __expf(S[n][e] - mnew);
        rs += S[n][e];
      }
#pragma unroll
      for (int o = 8; o >= 1; o >>= 1) rs += __shfl_xor(rs, o, 16);
      lrow[e] = lrow[e] * alpha + rs;
#pragma unroll
      for (int n = 0; n < 4; ++n) acc[n][e] = acc[n][e] * alpha;
    }

    // ---- P -> per-wave LDS as bf16 (same-wave LDS ops are in-order) ----
    unsigned short* Pw = &Ps[wave][0];
    const int mloc = (lane >> 4) * 8;
#pragma unroll
    for (int n = 0; n < 4; ++n)
#pragma unroll
      for (int e = 0; e < 8; ++e)
        Pw[(mloc + e) * 64 + n * 16 + (lane & 15)] = f2bf(S[n][e]);

    // ---- O += P V  (A = P from LDS, B = V^T rows d, k=j contiguous) ----
#pragma unroll
    for (int ks = 0; ks < 2; ++ks) {
      v16bf a = load_a(Pw + ks * 32, 64, lane);
#pragma unroll
      for (int n = 0; n < 4; ++n) {
        v16bf bv = load_b(&Vt[(n * 16) * 64 + ks * 32], 64, lane);
        acc[n] = wmma_bf16(a, bv, acc[n]);
      }
    }
  }

  // ---- normalize + store ctx (bf16, [b*T+t, h*64+d]) ----
#pragma unroll
  for (int e = 0; e < 8; ++e) {
    float rinv = 1.f / lrow[e];
    size_t m = rowQ + (lane >> 4) * 8 + e;
#pragma unroll
    for (int n = 0; n < 4; ++n) {
      ctx[m * Dc + h * DHc + n * 16 + (lane & 15)] = f2bf(acc[n][e] * rinv);
    }
  }
}

// ---------------------------------------------------------------------------
extern "C" void kernel_launch(void* const* d_in, const int* in_sizes, int n_in,
                              void* d_out, int out_size, void* d_ws, size_t ws_size,
                              hipStream_t stream) {
  (void)in_sizes; (void)n_in; (void)out_size; (void)ws_size;
  const float* x      = (const float*)d_in[0];
  const float* p      = (const float*)d_in[1];
  const float* s      = (const float*)d_in[2];
  const float* amask  = (const float*)d_in[3];
  const float* Wq     = (const float*)d_in[4];
  const float* Wk     = (const float*)d_in[5];
  const float* Wv     = (const float*)d_in[6];
  const float* Wo     = (const float*)d_in[7];
  const float* bo     = (const float*)d_in[8];
  const float* compat = (const float*)d_in[9];
  const float* gamma  = (const float*)d_in[10];

  // workspace carving (~60 MB)
  const size_t SZ_ACT = (size_t)Mtot * Dc * 2;     // 8 MiB bf16 activation
  const size_t SZ_W   = (size_t)Dc * Dc * 2;       // 2 MiB bf16 weight
  char* w = (char*)d_ws;
  unsigned short* xb   = (unsigned short*)w; w += SZ_ACT;
  unsigned short* wqb  = (unsigned short*)w; w += SZ_W;
  unsigned short* wkb  = (unsigned short*)w; w += SZ_W;
  unsigned short* wvb  = (unsigned short*)w; w += SZ_W;
  unsigned short* wob  = (unsigned short*)w; w += SZ_W;
  unsigned short* Qb   = (unsigned short*)w; w += SZ_ACT;
  unsigned short* Kb   = (unsigned short*)w; w += SZ_ACT;
  unsigned short* Vb   = (unsigned short*)w; w += SZ_ACT;
  unsigned short* Vtg  = (unsigned short*)w; w += SZ_ACT;  // V^T [B,H,DH,T]
  unsigned short* ctxb = (unsigned short*)w; w += SZ_ACT;
  float*          pcb  = (float*)w;          // [B,H,T,8] = 2 MiB

  // 1) converts
  {
    int n4 = (Mtot * Dc) / 4;                            // x: 4 Mi elems
    cvt_bf16x4<<<(n4 + 255) / 256, 256, 0, stream>>>(x, xb, n4);
    int w4 = (Dc * Dc) / 4;                              // each W: 1 Mi elems
    cvt_bf16x4<<<(w4 + 255) / 256, 256, 0, stream>>>(Wq, wqb, w4);
    cvt_bf16x4<<<(w4 + 255) / 256, 256, 0, stream>>>(Wk, wkb, w4);
    cvt_bf16x4<<<(w4 + 255) / 256, 256, 0, stream>>>(Wv, wvb, w4);
    cvt_bf16x4<<<(w4 + 255) / 256, 256, 0, stream>>>(Wo, wob, w4);
  }
  // 2) p_compat
  pc_kernel<<<(Bc * Hc * Tc * Kc) / 256, 256, 0, stream>>>(p, compat, pcb);
  // 3) QKV projections (one kernel, grid.z selects the weight)
  gemm_qkv<<<dim3(Dc / 128, Mtot / 128, 3), 256, 0, stream>>>(
      xb, wqb, wkb, wvb, Qb, Kb, Vb);
  // 4) one-time V transpose -> [B,H,DH,T]
  vtrans_kernel<<<dim3(Tc / 64, Hc, Bc), 256, 0, stream>>>(Vb, Vtg);
  // 5) flash attention
  attn_kernel<<<dim3(Tc / 128, Hc, Bc), 256, 0, stream>>>(
      Qb, Kb, Vtg, pcb, p, s, amask, gamma, ctxb);
  // 6) output projection + bias (f32 out)
  gemm_out<<<dim3(Dc / 128, Mtot / 128), 256, 0, stream>>>(
      ctxb, wob, bo, (float*)d_out);
}
